// RawSizeRoutedDepthModel_33689723469844
// MI455X (gfx1250) — matrix operand order repaired and checked by reading, not compiled, not run on hardware
//
#include <hip/hip_runtime.h>

typedef _Float16 v16h __attribute__((ext_vector_type(16)));
typedef __fp16   v2fp __attribute__((ext_vector_type(2)));
typedef float    v8f  __attribute__((ext_vector_type(8)));

#define NE       7
#define DIM      64
#define ROWS_WG  512
#define TPB      256
#define MAXT     40
#define LISTSZ   640

// ---------- pre-kernel: transpose expert weights to f16 A-fragment-friendly layout ----------
// ws layout (f16):  W1t [7][64][64]  (out-ch-major rows of k)  at 0
//                   W2t [7][64][64]                             at 28672
//                   W3t [7][16][64]  (rows 3..15 zeroed)        at 57344
// total 64512 f16 = 129024 bytes of d_ws
__global__ __launch_bounds__(256) void prep_weights(const float* __restrict__ W1,
                                                    const float* __restrict__ W2,
                                                    const float* __restrict__ W3,
                                                    _Float16* __restrict__ wt) {
  int i = blockIdx.x * 256 + threadIdx.x;
  if (i < NE * 4096) {
    int e = i >> 12, c = (i >> 6) & 63, k = i & 63;
    wt[i] = (_Float16)W1[(e << 12) + (k << 6) + c];
  } else if (i < 2 * NE * 4096) {
    int j = i - NE * 4096;
    int e = j >> 12, c = (j >> 6) & 63, k = j & 63;
    wt[i] = (_Float16)W2[(e << 12) + (k << 6) + c];
  } else if (i < 2 * NE * 4096 + NE * 1024) {
    int j = i - 2 * NE * 4096;
    int e = j >> 10, c = (j >> 6) & 15, k = j & 63;
    wt[i] = (c < 3) ? (_Float16)W3[e * 192 + k * 3 + c] : (_Float16)0.0f;
  }
}

// Load one A fragment (16x32 f16): halves 0..7 at p, halves 8..15 at p+8 elements.
static __device__ __forceinline__ v16h loadA(const _Float16* p) {
  union { uint4 u[2]; v16h v; } t;
  t.u[0] = *(const uint4*)(p);
  t.u[1] = *(const uint4*)(p + 8);
  return t.v;
}

// Pack two f32 into one dword of two f16: exactly one v_cvt_pk_rtz_f16_f32,
// pairing fixed at the IR level so the compiler cannot re-shuffle with v_perm_b32.
static __device__ __forceinline__ unsigned int pk16(float a, float b) {
  union { v2fp h; unsigned int u; } t;
  t.h = __builtin_amdgcn_cvt_pkrtz(a, b);
  return t.u;
}

// One 64->64 layer: D = Wt_tile x B + bias(C), then ReLU.  Emits 8 v_wmma per call.
static __device__ __forceinline__ void layer64(const _Float16* __restrict__ Wt,
                                               const float* __restrict__ b, int e,
                                               int nlo, int kbA, bool hiH,
                                               v16h B0, v16h B1, v8f acc[4]) {
#pragma unroll
  for (int t = 0; t < 4; ++t) {
    const float* bp = b + e * DIM + t * 16 + (hiH ? 8 : 0);
    float4 c0 = *(const float4*)(bp);
    float4 c1 = *(const float4*)(bp + 4);
    v8f c;
    c[0] = c0.x; c[1] = c0.y; c[2] = c0.z; c[3] = c0.w;
    c[4] = c1.x; c[5] = c1.y; c[6] = c1.z; c[7] = c1.w;
    const _Float16* wrow = Wt + e * 4096 + (t * 16 + nlo) * 64;
    c = __builtin_amdgcn_wmma_f32_16x16x32_f16(false, loadA(wrow + kbA),      false, B0,
                                               (short)0, c, false, false);
    c = __builtin_amdgcn_wmma_f32_16x16x32_f16(false, loadA(wrow + 32 + kbA), false, B1,
                                               (short)0, c, false, false);
#pragma unroll
    for (int r = 0; r < 8; ++r) c[r] = fmaxf(c[r], 0.0f);
    acc[t] = c;
  }
}

// Repack 16x64 f32 D (channels-as-M) into two f16 B fragments for the next layer.
// Pack to f16 pairs FIRST, then half-wave-swap packed dwords: 16 cvt_pk + 16
// shuffles + 16 b32 selects per layer.
static __device__ __forceinline__ void relayout(const v8f a[4], bool hiH, v16h& F0, v16h& F1) {
  unsigned int p[4][4];
#pragma unroll
  for (int t = 0; t < 4; ++t)
#pragma unroll
    for (int j = 0; j < 4; ++j) p[t][j] = pk16(a[t][2 * j], a[t][2 * j + 1]);

  union { unsigned int u[8]; v16h v; } f0, f1;
#pragma unroll
  for (int j = 0; j < 4; ++j) {
    unsigned int s0 = __shfl_xor((int)p[0][j], 16, 32);
    unsigned int s1 = __shfl_xor((int)p[1][j], 16, 32);
    unsigned int s2 = __shfl_xor((int)p[2][j], 16, 32);
    unsigned int s3 = __shfl_xor((int)p[3][j], 16, 32);
    f0.u[j]     = hiH ? s1      : p[0][j];   // ch 0..7   | 16..23
    f0.u[4 + j] = hiH ? p[1][j] : s0;        // ch 8..15  | 24..31
    f1.u[j]     = hiH ? s3      : p[2][j];   // ch 32..39 | 48..55
    f1.u[4 + j] = hiH ? p[3][j] : s2;        // ch 40..47 | 56..63
  }
  F0 = f0.v;
  F1 = f1.v;
}

__global__ __launch_bounds__(TPB) void moe_fwd(const float* __restrict__ x,
                                               const int* __restrict__ sidx,
                                               const float* __restrict__ b1,
                                               const float* __restrict__ b2,
                                               const float* __restrict__ b3,
                                               const _Float16* __restrict__ wt,
                                               float* __restrict__ out, int n) {
  __shared__ int cnt[NE], cnt2[NE], off_[NE], total_s;
  __shared__ short tile_e[MAXT];
  __shared__ unsigned short list_[LISTSZ];

  const int tid  = threadIdx.x;
  const int base = blockIdx.x * ROWS_WG;
  int rows = n - base; if (rows > ROWS_WG) rows = ROWS_WG;

  // ---- bucket rows of this block by expert into 16-row tiles ----
  if (tid < NE) { cnt[tid] = 0; cnt2[tid] = 0; }
  for (int i = tid; i < LISTSZ; i += TPB) list_[i] = 0xFFFFu;
  __syncthreads();
  for (int r = tid; r < rows; r += TPB) atomicAdd(&cnt[sidx[base + r]], 1);
  __syncthreads();
  if (tid == 0) {
    int a = 0, t = 0;
    for (int e = 0; e < NE; ++e) {
      off_[e] = a;
      int nt = (cnt[e] + 15) >> 4;
      for (int j = 0; j < nt; ++j) tile_e[t + j] = (short)e;
      t += nt; a += nt << 4;
    }
    total_s = t;
  }
  __syncthreads();
  for (int r = tid; r < rows; r += TPB) {
    int e = sidx[base + r];
    int pos = off_[e] + atomicAdd(&cnt2[e], 1);
    list_[pos] = (unsigned short)r;
  }
  __syncthreads();

  const int  lane = tid & 31;
  const int  wv   = tid >> 5;
  const int  nlo  = lane & 15;
  const bool hiH  = lane >= 16;
  const int  kbB  = hiH ? 16 : 0;   // B-frag K base within 32-chunk
  const int  kbA  = hiH ? 16 : 0;   // A-frag K base within 32-chunk
  const int  total = total_s;

  const _Float16* W1t = wt;
  const _Float16* W2t = wt + NE * 4096;
  const _Float16* W3t = wt + 2 * NE * 4096;

  for (int ti = wv; ti < total; ti += TPB / 32) {
    const int e = tile_e[ti];
    const unsigned short rl = list_[ti * 16 + nlo];
    const bool valid = (rl != (unsigned short)0xFFFF);
    const int  grow  = base + (valid ? (int)rl : 0);

    // ---- activations -> B fragments (x^T: K = input channel, N = batch row) ----
    // Pad slots alias row 0; their (finite) garbage stays in their own column and
    // is never stored, so no masking of the data is needed.
    const float* xr = x + (size_t)grow * DIM;
    union { float4 f4[4]; float f[16]; } u0, u1;
    u0.f4[0] = *(const float4*)(xr + kbB + 0);
    u0.f4[1] = *(const float4*)(xr + kbB + 4);
    u0.f4[2] = *(const float4*)(xr + kbB + 8);
    u0.f4[3] = *(const float4*)(xr + kbB + 12);
    u1.f4[0] = *(const float4*)(xr + 32 + kbB + 0);
    u1.f4[1] = *(const float4*)(xr + 32 + kbB + 4);
    u1.f4[2] = *(const float4*)(xr + 32 + kbB + 8);
    u1.f4[3] = *(const float4*)(xr + 32 + kbB + 12);
    union { unsigned int u[8]; v16h v; } b0p, b1p;
#pragma unroll
    for (int j = 0; j < 8; ++j) {
      b0p.u[j] = pk16(u0.f[2 * j], u0.f[2 * j + 1]);
      b1p.u[j] = pk16(u1.f[2 * j], u1.f[2 * j + 1]);
    }
    v16h B0 = b0p.v, B1 = b1p.v;

    // ---- layer 1 & 2 (bias folded into C, ReLU on f32 acc) ----
    v8f h1[4], h2[4];
    layer64(W1t, b1, e, nlo, kbA, hiH, B0, B1, h1);
    v16h H0, H1;
    relayout(h1, hiH, H0, H1);
    layer64(W2t, b2, e, nlo, kbA, hiH, H0, H1, h2);
    v16h G0, G1;
    relayout(h2, hiH, G0, G1);

    // ---- layer 3: 64 -> 3 (padded to 16 output rows) ----
    v8f a3;
#pragma unroll
    for (int r = 0; r < 8; ++r) a3[r] = 0.0f;
#pragma unroll
    for (int r = 0; r < 3; ++r) {
      float bv = b3[e * 3 + r];
      a3[r] = hiH ? 0.0f : bv;
    }
    const _Float16* w3row = W3t + e * 1024 + nlo * 64;
    a3 = __builtin_amdgcn_wmma_f32_16x16x32_f16(false, loadA(w3row + kbA),      false, G0,
                                                (short)0, a3, false, false);
    a3 = __builtin_amdgcn_wmma_f32_16x16x32_f16(false, loadA(w3row + 32 + kbA), false, G1,
                                                (short)0, a3, false, false);

    // ---- softmax + store (logits in acc VGPRs 0..2 of low half-wave) ----
    if (!hiH && valid) {
      int row = base + (int)rl;
      float l0 = a3[0], l1 = a3[1], l2 = a3[2];
      float m  = fmaxf(l0, fmaxf(l1, l2));
      float e0 = __expf(l0 - m), e1 = __expf(l1 - m), e2 = __expf(l2 - m);
      float inv = 1.0f / (e0 + e1 + e2);
      float* lo = out + (size_t)row * 3;
      lo[0] = l0; lo[1] = l1; lo[2] = l2;
      float* po = out + (size_t)n * 3 + (size_t)row * 3;
      po[0] = e0 * inv; po[1] = e1 * inv; po[2] = e2 * inv;
    }
  }
}

extern "C" void kernel_launch(void* const* d_in, const int* in_sizes, int n_in,
                              void* d_out, int out_size, void* d_ws, size_t ws_size,
                              hipStream_t stream) {
  const float* x    = (const float*)d_in[0];
  const int*   sidx = (const int*)d_in[1];
  const float* W1   = (const float*)d_in[2];
  const float* b1   = (const float*)d_in[3];
  const float* W2   = (const float*)d_in[4];
  const float* b2   = (const float*)d_in[5];
  const float* W3   = (const float*)d_in[6];
  const float* b3   = (const float*)d_in[7];
  float* out = (float*)d_out;
  _Float16* wt = (_Float16*)d_ws;   // needs 129024 bytes
  int n = in_sizes[1];

  const int prep_elems = 2 * NE * 4096 + NE * 1024;  // 64512
  hipLaunchKernelGGL(prep_weights, dim3((prep_elems + 255) / 256), dim3(256), 0, stream,
                     W1, W2, W3, wt);
  hipLaunchKernelGGL(moe_fwd, dim3((n + ROWS_WG - 1) / ROWS_WG), dim3(TPB), 0, stream,
                     x, sidx, b1, b2, b3, (const _Float16*)wt, out, n);
}
